// ESCAPE_NET_SNN_STDB_77781857730950
// MI455X (gfx1250) — matrix operand order repaired and checked by reading, not compiled, not run on hardware
//
#include <hip/hip_runtime.h>
#include <hip/hip_bf16.h>

typedef __attribute__((ext_vector_type(16))) __bf16 v16bf;
typedef __attribute__((ext_vector_type(8)))  float  v8f;

// Compile-time offset of flattened k-index c within a zero-padded [CIN][HP][WP]
// input image: c = (ci << 2*KLOG) | (ky << KLOG) | kx  (bitfields are disjoint,
// so delta() is additive over disjoint-bit summands).
template <int KLOG, int HP, int WP>
__host__ __device__ constexpr int conv_delta(int c) {
  constexpr int KSZ = 1 << KLOG;
  int kx = c & (KSZ - 1);
  int ky = (c >> KLOG) & (KSZ - 1);
  int ci = c >> (2 * KLOG);
  return (ci * HP + ky) * WP + kx;
}

// ---------------------------------------------------------------- RNG (threefry2x32)
__device__ __forceinline__ unsigned rotl32(unsigned v, unsigned n) {
  return (v << n) | (v >> (32u - n));
}

__device__ __forceinline__ void threefry2x32(unsigned k0, unsigned k1,
                                             unsigned c0, unsigned c1,
                                             unsigned& o0, unsigned& o1) {
  const unsigned ks2 = k0 ^ k1 ^ 0x1BD11BDAu;
  unsigned x0 = c0 + k0, x1 = c1 + k1;
  const unsigned rA[4] = {13u, 15u, 26u, 6u};
  const unsigned rB[4] = {17u, 29u, 16u, 24u};
#pragma unroll
  for (int i = 0; i < 4; i++) { x0 += x1; x1 = rotl32(x1, rA[i]); x1 ^= x0; }
  x0 += k1; x1 += ks2 + 1u;
#pragma unroll
  for (int i = 0; i < 4; i++) { x0 += x1; x1 = rotl32(x1, rB[i]); x1 ^= x0; }
  x0 += ks2; x1 += k0 + 2u;
#pragma unroll
  for (int i = 0; i < 4; i++) { x0 += x1; x1 = rotl32(x1, rA[i]); x1 ^= x0; }
  x0 += k0; x1 += k1 + 3u;
#pragma unroll
  for (int i = 0; i < 4; i++) { x0 += x1; x1 = rotl32(x1, rB[i]); x1 ^= x0; }
  x0 += k1; x1 += ks2 + 4u;
#pragma unroll
  for (int i = 0; i < 4; i++) { x0 += x1; x1 = rotl32(x1, rA[i]); x1 ^= x0; }
  x0 += ks2; x1 += k0 + 5u;
  o0 = x0; o1 = x1;
}

// ---------------------------------------------------------------- Poisson spike input -> padded [8][63][107]
__global__ __launch_bounds__(256) void gen_input(const float* __restrict__ x,
                                                 __bf16* __restrict__ inp_p,
                                                 int t, int n) {
  int i = blockIdx.x * blockDim.x + threadIdx.x;
  if (i >= n) return;
  unsigned k0, k1, r0, r1;
  threefry2x32(0u, 42u, 0u, (unsigned)t, k0, k1);       // fold_in(key(42), t)
  threefry2x32(k0, k1, 0u, (unsigned)i, r0, r1);        // per-element counter
  float u = (float)(r0 >> 8) * (1.0f / 16777216.0f);    // [0,1)
  float xv = x[i];
  float sgn = (xv > 0.f) ? 1.f : ((xv < 0.f) ? -1.f : 0.f);
  float s = (u <= fabsf(xv) * 0.9f) ? sgn : 0.f;
  int xq = i % 100; int tm = i / 100; int y = tm % 56; int b = tm / 56;
  inp_p[(b * 63 + (y + 3)) * 107 + (xq + 3)] = (__bf16)s;   // interior only
}

// ---------------------------------------------------------------- implicit-GEMM conv (bf16 WMMA)
// One wave -> 16 M pixels x 64 N out-channels (4 accumulators, 4 WMMA per k-step).
// Weight panels double-buffered in LDS, staged cooperatively by the whole block
// (each 32x64 bf16 panel loaded once per block, not once per wave).
// Input is zero-padded [CIN][HP][WP]: all A loads unconditional, at compile-time
// offsets from a per-lane base; per-k-step advance is a constant stride.
template <int KLOG, int CIN, int H, int W, int HP, int WP, int MTOT, int WAVES>
__global__ __launch_bounds__(WAVES * 32) void conv_wmma_t(const __bf16* __restrict__ in,
                                                          const __bf16* __restrict__ wB,
                                                          float* __restrict__ drive) {
  constexpr int KTOT  = CIN * (1 << KLOG) * (1 << KLOG);
  constexpr int NSTEP = KTOT / 32;
  constexpr int STEP  = conv_delta<KLOG, HP, WP>(32);   // constant per-k-step stride
  constexpr int NTHR  = WAVES * 32;
  static_assert((MTOT >> 4) % WAVES == 0, "grid must be exact (no exited waves)");
  __shared__ __bf16 lbuf[2][2048];                      // 2 x 4KB weight panels

  int tid  = threadIdx.x;
  int lane = tid & 31;
  int wave = blockIdx.x * WAVES + (tid >> 5);           // == m-tile

  int m  = (wave << 4) + (lane & 15);
  int xx = m % W; int tm = m / W; int yy = tm % H; int bb = tm / H;  // const-div
  const __bf16* ap = in + ((bb * CIN) * HP + yy) * WP + xx +
                     ((lane >= 16) ? conv_delta<KLOG, HP, WP>(8) : 0);

  auto stage = [&](int s) {                             // stage panel s -> lbuf[s&1]
    const __bf16* wp = wB + s * 2048;
    for (int c = tid * 8; c < 2048; c += NTHR * 8) {
#pragma unroll
      for (int e = 0; e < 8; e++) lbuf[s & 1][c + e] = wp[c + e];
    }
  };

  stage(0);
  v8f acc[4] = {};
  for (int s = 0; s < NSTEP; s++) {
    __syncthreads();                                    // panel s visible; old readers done
    if (s + 1 < NSTEP) {
      __builtin_prefetch((const void*)(wB + (s + 2) * 2048), 0, 3);
      stage(s + 1);
    }
    v16bf bf[4];
#pragma unroll
    for (int nt = 0; nt < 4; nt++)
#pragma unroll
      for (int j = 0; j < 16; j++) bf[nt][j] = lbuf[s & 1][(((j < 8) ? j : (8 + j)) + ((lane >> 4) << 3)) * 64 + nt * 16 + (lane & 15)];
    v16bf a;
#pragma unroll
    for (int j = 0; j < 16; j++)
      a[j] = ap[conv_delta<KLOG, HP, WP>((j < 8) ? j : (16 + (j - 8)))];
    ap += STEP;
#pragma unroll
    for (int nt = 0; nt < 4; nt++)
      acc[nt] = __builtin_amdgcn_wmma_f32_16x16x32_bf16(
          false, a, false, bf[nt], (short)0, acc[nt], false, false);
  }

  // C/D layout: lane n = lane&15, VGPR r -> M = r + 8*(lane>=16)
  int nl    = lane & 15;
  int mrow0 = (wave << 4) + ((lane >= 16) ? 8 : 0);
#pragma unroll
  for (int r = 0; r < 8; r++) {
    int mm = mrow0 + r;
    int ox = mm % W; int t2 = mm / W; int oy = t2 % H; int ob = t2 / H;
    size_t base = (((size_t)ob * 64) * H + oy) * W + ox;
#pragma unroll
    for (int nt = 0; nt < 4; nt++)
      drive[base + (size_t)(nt * 16 + nl) * (H * W)] = acc[nt][r];
  }
}

// ---------------------------------------------------------------- LIF (+2x2 avgpool of spikes)
// TH=1, LEAK=1: s = (mem_old > 1); mem = mem_old + drive - s   (spike from OLD membrane)
template <int H, int W, int HP, int WP, int PADLO>
__global__ __launch_bounds__(256) void lif_pool(const float* __restrict__ drive,
                                                float* __restrict__ mem,
                                                __bf16* __restrict__ pooled_p, int npool) {
  int i = blockIdx.x * blockDim.x + threadIdx.x;
  if (i >= npool) return;
  constexpr int PW = W >> 1, PH = H >> 1;
  int px = i % PW; int t = i / PW; int py = t % PH; int bc = t / PH;  // bc = b*64+c
  float sum = 0.f;
#pragma unroll
  for (int dy = 0; dy < 2; dy++)
#pragma unroll
    for (int dx = 0; dx < 2; dx++) {
      size_t off = ((size_t)bc * H + (py * 2 + dy)) * W + (px * 2 + dx);
      float mo = mem[off];
      float s  = (mo > 1.0f) ? 1.0f : 0.0f;
      mem[off] = mo + drive[off] - s;
      sum += s;
    }
  pooled_p[((size_t)bc * HP + (py + PADLO)) * WP + (px + PADLO)] = (__bf16)(sum * 0.25f);
}

__global__ __launch_bounds__(256) void lif_flat(const float* __restrict__ drive,
                                                float* __restrict__ mem,
                                                __bf16* __restrict__ spk, int n) {
  int i = blockIdx.x * blockDim.x + threadIdx.x;
  if (i >= n) return;
  float mo = mem[i];
  float s  = (mo > 1.0f) ? 1.0f : 0.0f;
  mem[i] = mo + drive[i] - s;
  spk[i] = (__bf16)s;                      // rows 0..7 of 16-row padded A matrix
}

// ---------------------------------------------------------------- FC1: 16(M, 8 valid+8 zero rows) x 256(N), K=22400
// One block = one K chunk (640 K, 20 steps) x 4 waves (one 64-N group each).
// 32x256 weight panel (16KB) double-buffered in LDS, shared by all 4 groups.
__global__ __launch_bounds__(128) void fc1_wmma(const __bf16* __restrict__ s3p,
                                                const __bf16* __restrict__ wB,
                                                float* __restrict__ part) {
  __shared__ __bf16 lbuf[2][8192];                      // 2 x 16KB
  int tid  = threadIdx.x;
  int lane = tid & 31;
  int group = tid >> 5;                                 // 0..3 : which 64 of 256 N
  int chunk = blockIdx.x;                               // 0..34
  int kstart = chunk * 640;
  const __bf16* ap0 = s3p + (size_t)(lane & 15) * 22400 + ((lane >= 16) ? 8 : 0);

  auto stage = [&](int s) {                             // stage step-s panel
    const __bf16* wp = wB + (size_t)(kstart + s * 32) * 256;
    for (int c = tid * 16; c < 8192; c += 128 * 16) {
#pragma unroll
      for (int e = 0; e < 16; e++) lbuf[s & 1][c + e] = wp[c + e];
    }
  };

  stage(0);
  v8f acc[4] = {};
  for (int s = 0; s < 20; s++) {
    __syncthreads();
    if (s + 1 < 20) {
      __builtin_prefetch((const void*)(wB + (size_t)(kstart + (s + 2) * 32) * 256), 0, 3);
      stage(s + 1);
    }
    v16bf bf[4];
#pragma unroll
    for (int nt = 0; nt < 4; nt++)
#pragma unroll
      for (int j = 0; j < 16; j++)
        bf[nt][j] = lbuf[s & 1][(((j < 8) ? j : (8 + j)) + ((lane >> 4) << 3)) * 256 + group * 64 + nt * 16 + (lane & 15)];
    const __bf16* ap = ap0 + kstart + s * 32;
    v16bf a;
#pragma unroll
    for (int j = 0; j < 8; j++) { a[j] = ap[j]; a[8 + j] = ap[16 + j]; }
#pragma unroll
    for (int nt = 0; nt < 4; nt++)
      acc[nt] = __builtin_amdgcn_wmma_f32_16x16x32_bf16(
          false, a, false, bf[nt], (short)0, acc[nt], false, false);
  }
  int nl = lane & 15;
  int mb = (lane >= 16) ? 8 : 0;
#pragma unroll
  for (int nt = 0; nt < 4; nt++) {
    int n = group * 64 + nt * 16 + nl;
#pragma unroll
    for (int r = 0; r < 8; r++)
      part[chunk * 4096 + n * 16 + mb + r] = acc[nt][r];
  }
}

// ---------------------------------------------------------------- split-K reduce + FC1 LIF -> FC2 -> LIF -> softmax
__global__ __launch_bounds__(256) void fc_tail(const float* __restrict__ part,
                                               float* __restrict__ mf1,
                                               const float* __restrict__ wf2,
                                               float* __restrict__ mf2,
                                               float* __restrict__ out) {
  __shared__ float sf1[8 * 256];
  int j = threadIdx.x;                      // n index
  for (int b = 0; b < 8; b++) {
    float d = 0.f;
#pragma unroll
    for (int c = 0; c < 35; c++) d += part[c * 4096 + j * 16 + b];
    float mo = mf1[b * 256 + j];
    float s  = (mo > 1.0f) ? 1.0f : 0.0f;
    mf1[b * 256 + j] = mo + d - s;
    sf1[b * 256 + j] = s;
  }
  __syncthreads();
  if (j < 8) {
    int b = j;
    float d[3];
    for (int o = 0; o < 3; o++) {
      float a = 0.f;
      for (int q = 0; q < 256; q++) a += sf1[b * 256 + q] * wf2[o * 256 + q];
      d[o] = a;
    }
    float sp[3];
    for (int o = 0; o < 3; o++) {
      float mo = mf2[b * 3 + o];
      float s  = (mo > 1.0f) ? 1.0f : 0.0f;
      mf2[b * 3 + o] = mo + d[o] - s;
      sp[o] = s;
    }
    float mx = fmaxf(sp[0], fmaxf(sp[1], sp[2]));
    float e0 = __expf(sp[0] - mx), e1 = __expf(sp[1] - mx), e2 = __expf(sp[2] - mx);
    float inv = 1.f / (e0 + e1 + e2);
    mf2[b * 3 + 0] += e0 * inv;
    mf2[b * 3 + 1] += e1 * inv;
    mf2[b * 3 + 2] += e2 * inv;
    out[b * 3 + 0] = mf2[b * 3 + 0];
    out[b * 3 + 1] = mf2[b * 3 + 1];
    out[b * 3 + 2] = mf2[b * 3 + 2];
  }
}

// ---------------------------------------------------------------- weight prep (f32 -> bf16, [K][N] layout)
__global__ __launch_bounds__(256) void prep_w(const float* __restrict__ w,
                                              __bf16* __restrict__ wB, int Ktot) {
  int tid = blockIdx.x * blockDim.x + threadIdx.x;
  if (tid >= Ktot * 64) return;
  int o = tid & 63, k = tid >> 6;                 // w[o][ci][ky][kx] flat = o*Ktot + k
  wB[(size_t)k * 64 + o] = (__bf16)w[(size_t)o * Ktot + k];
}

__global__ __launch_bounds__(256) void prep_fc1w(const float* __restrict__ w,
                                                 __bf16* __restrict__ wB) {
  int tid = blockIdx.x * blockDim.x + threadIdx.x;
  if (tid >= 22400 * 256) return;
  int n = tid & 255, k = tid >> 8;
  wB[(size_t)k * 256 + n] = (__bf16)w[(size_t)n * 22400 + k];
}

// ---------------------------------------------------------------- launch
extern "C" void kernel_launch(void* const* d_in, const int* in_sizes, int n_in,
                              void* d_out, int out_size, void* d_ws, size_t ws_size,
                              hipStream_t stream) {
  (void)in_sizes; (void)n_in; (void)out_size; (void)ws_size;
  const float* x   = (const float*)d_in[0];
  const float* w1  = (const float*)d_in[1];
  const float* w2  = (const float*)d_in[2];
  const float* w3  = (const float*)d_in[3];
  const float* wf1 = (const float*)d_in[4];
  const float* wf2 = (const float*)d_in[5];
  float* out = (float*)d_out;

  char* ws = (char*)d_ws;
  size_t off = 0;
  auto alloc = [&](size_t bytes) -> char* {
    char* p = ws + off;
    off += (bytes + 255) & ~(size_t)255;
    return p;
  };
  __bf16* w1b  = (__bf16*)alloc((size_t)64 * 64 * 2);
  __bf16* w2b  = (__bf16*)alloc((size_t)1024 * 64 * 2);
  __bf16* w3b  = (__bf16*)alloc((size_t)256 * 64 * 2);
  __bf16* wf1b = (__bf16*)alloc((size_t)22400 * 256 * 2);
  // zero-padded conv inputs (halo zeroed once; producers write interiors only)
  const size_t inp_sz = (size_t)8 * 63 * 107 * 2;            // pad 3/4 for k=8
  const size_t p1_sz  = (size_t)8 * 64 * 31 * 53 * 2;        // pad 1/2 for k=4
  const size_t p2_sz  = (size_t)8 * 64 * 15 * 26 * 2;        // pad 0/1 for k=2
  const size_t s3_sz  = (size_t)16 * 22400 * 2;              // M padded 8 -> 16
  __bf16* inp_p = (__bf16*)alloc(inp_sz);
  __bf16* p1_p  = (__bf16*)alloc(p1_sz);
  __bf16* p2_p  = (__bf16*)alloc(p2_sz);
  float* drive1 = (float*)alloc((size_t)8 * 64 * 56 * 100 * 4);
  float* m1     = (float*)alloc((size_t)8 * 64 * 56 * 100 * 4);
  float* drive2 = (float*)alloc((size_t)8 * 64 * 28 * 50 * 4);
  float* m2     = (float*)alloc((size_t)8 * 64 * 28 * 50 * 4);
  float* drive3 = (float*)alloc((size_t)8 * 64 * 14 * 25 * 4);
  float* m3     = (float*)alloc((size_t)8 * 64 * 14 * 25 * 4);
  __bf16* s3p   = (__bf16*)alloc(s3_sz);
  float* part   = (float*)alloc((size_t)35 * 4096 * 4);
  float* df_un  = (float*)alloc((size_t)2048 * 4);  (void)df_un; // spare
  float* mf1    = (float*)alloc((size_t)2048 * 4);
  float* mf2    = (float*)alloc((size_t)24 * 4);

  // zero membranes + padded halos (graph-capture-safe, once per launch)
  hipMemsetAsync(m1,  0, (size_t)8 * 64 * 56 * 100 * 4, stream);
  hipMemsetAsync(m2,  0, (size_t)8 * 64 * 28 * 50 * 4, stream);
  hipMemsetAsync(m3,  0, (size_t)8 * 64 * 14 * 25 * 4, stream);
  hipMemsetAsync(mf1, 0, (size_t)2048 * 4, stream);
  hipMemsetAsync(mf2, 0, (size_t)24 * 4, stream);
  hipMemsetAsync(inp_p, 0, inp_sz, stream);
  hipMemsetAsync(p1_p,  0, p1_sz, stream);
  hipMemsetAsync(p2_p,  0, p2_sz, stream);
  hipMemsetAsync(s3p,   0, s3_sz, stream);

  // weight prep
  prep_w<<<(64 * 64 + 255) / 256, 256, 0, stream>>>(w1, w1b, 64);
  prep_w<<<(1024 * 64 + 255) / 256, 256, 0, stream>>>(w2, w2b, 1024);
  prep_w<<<(256 * 64 + 255) / 256, 256, 0, stream>>>(w3, w3b, 256);
  prep_fc1w<<<22400, 256, 0, stream>>>(wf1, wf1b);

  for (int t = 0; t < 100; t++) {
    gen_input<<<175, 256, 0, stream>>>(x, inp_p, t, 44800);
    // conv1: k=8, K=64, M=44800 -> 2800 waves, 8 waves/block
    conv_wmma_t<3, 1, 56, 100, 63, 107, 44800, 8><<<350, 256, 0, stream>>>(inp_p, w1b, drive1);
    lif_pool<56, 100, 31, 53, 1><<<2800, 256, 0, stream>>>(drive1, m1, p1_p, 716800);
    // conv2: k=4, K=1024, M=11200 -> 700 waves, 4 waves/block
    conv_wmma_t<2, 64, 28, 50, 31, 53, 11200, 4><<<175, 128, 0, stream>>>(p1_p, w2b, drive2);
    lif_pool<28, 50, 15, 26, 0><<<700, 256, 0, stream>>>(drive2, m2, p2_p, 179200);
    // conv3: k=2, K=256, M=2800 -> 175 waves, 5 waves/block
    conv_wmma_t<1, 64, 14, 25, 15, 26, 2800, 5><<<35, 160, 0, stream>>>(p2_p, w3b, drive3);
    lif_flat<<<700, 256, 0, stream>>>(drive3, m3, s3p, 179200);
    fc1_wmma<<<35, 128, 0, stream>>>(s3p, wf1b, part);
    fc_tail<<<1, 256, 0, stream>>>(part, mf1, wf2, mf2, out);
  }
}
